// GNNHist_83330955477524
// MI455X (gfx1250) — matrix operand order repaired and checked by reading, not compiled, not run on hardware
//
#include <hip/hip_runtime.h>
#include <hip/hip_bf16.h>

typedef __attribute__((ext_vector_type(16))) _Float16 v16h;
typedef __attribute__((ext_vector_type(8)))  float    v8f;

#define B_    256
#define U1_   1025
#define SLEN  1537
#define N_    (B_ * SLEN)      // 393472
#define D_    64
#define E_    2000000
#define FFIN  135
#define FFH   100
#define KPAD  160              // 5 k-steps of 32
#define NPAD  112              // 7 n-tiles of 16
#define MTOT  (B_ * U1_)       // 262400
#define MTILES (MTOT / 16)     // 16400 (exact)

// ---------------------------------------------------------------------------
// Kernel 1: fold the rank-1 GNN into four 64-vectors:
//   P  = W_in @ W_self,  Q = W_in @ W_nbr,  Qb = b_in @ W_nbr,
//   C0 = b_in @ W_self + b_gnn
// so emb[n,d] = relu(nf[n]*P[d] + sA[n]*Q[d] + sB[n]*Qb[d] + C0[d])
// ---------------------------------------------------------------------------
__global__ void prep_consts(const float* __restrict__ W_in, const float* __restrict__ b_in,
                            const float* __restrict__ W_self, const float* __restrict__ W_nbr,
                            const float* __restrict__ b_gnn, float* __restrict__ consts) {
    int d = threadIdx.x;  // 64 threads
    float p = 0.f, q = 0.f, qb = 0.f, c0 = 0.f;
    for (int k = 0; k < D_; ++k) {
        float wi = W_in[k], bi = b_in[k];
        p  += wi * W_self[k * D_ + d];
        q  += wi * W_nbr[k * D_ + d];
        qb += bi * W_nbr[k * D_ + d];
        c0 += bi * W_self[k * D_ + d];
    }
    consts[d]       = p;
    consts[64 + d]  = q;
    consts[128 + d] = qb;
    consts[192 + d] = c0 + b_gnn[d];
}

__global__ void zero_ws(float* __restrict__ p, int n) {
    int i = blockIdx.x * blockDim.x + threadIdx.x;
    if (i < n) p[i] = 0.f;
}

// per-row mean of w (B rows of U1)
__global__ void row_mean(const float* __restrict__ w, float* __restrict__ meanw) {
    __shared__ float red[256];
    int b = blockIdx.x, tid = threadIdx.x;
    float s = 0.f;
    for (int u = tid; u < U1_; u += 256) s += w[b * U1_ + u];
    red[tid] = s; __syncthreads();
    for (int o = 128; o > 0; o >>= 1) {
        if (tid < o) red[tid] += red[tid + o];
        __syncthreads();
    }
    if (tid == 0) meanw[b] = red[0] / (float)U1_;
}

// scalar segment sums over edges: sA[d] += w_e * nf[src], sB[d] += w_e
__global__ void edge_scatter(const int* __restrict__ src, const int* __restrict__ dst,
                             const float* __restrict__ wt, const float* __restrict__ nf,
                             float* __restrict__ sA, float* __restrict__ sB) {
    int e = blockIdx.x * blockDim.x + threadIdx.x;
    if (e >= E_) return;
    int s = src[e], d = dst[e];
    float we = wt[e];
    atomicAdd(&sA[d], we * nf[s]);
    atomicAdd(&sB[d], we);
}

// ---------------------------------------------------------------------------
// Fused feature-build + MLP via WMMA.
// One wave per 16-row M tile; 7 N-tiles x 5 K-steps of v_wmma_f32_16x16x32_f16.
// ---------------------------------------------------------------------------
struct FeatCtx {
    const float *w, *meanw, *hs, *hss, *hd, *size, *nf, *sA, *sB;
    const float *cP, *cQ, *cQb, *cC0;   // LDS-resident 64-vectors
    float fi;
    int b, u, node_emb, node_inc;
};

__device__ __forceinline__ float emb_val(const FeatCtx& c, int node, int d) {
    float v = c.nf[node] * c.cP[d] + c.sA[node] * c.cQ[d] + c.sB[node] * c.cQb[d] + c.cC0[d];
    return v > 0.f ? v : 0.f;
}

__device__ __forceinline__ float feat(const FeatCtx& c, int k) {
    if (k >= FFIN) return 0.f;                              // K padding
    if (k >= 71)   return emb_val(c, c.node_emb, k - 71);   // embeddings[:, u, :]
    if (k >= 7)    return emb_val(c, c.node_inc, k - 7);    // incoming
    int bu = c.b * U1_ + c.u;
    switch (k) {
        case 0: return c.w[bu];
        case 1: return c.meanw[c.b];
        case 2: return c.u == 0 ? -1.f : c.hs[bu] / c.fi;
        case 3: { if (c.u == 0) return -1.f;
                  float h = c.hs[bu];
                  return (c.hss[bu] - h * h / c.fi) / c.fi; }
        case 4: return c.u == 0 ? -1.f : c.hd[bu] / c.fi;
        case 5: return c.fi;
        default: return c.size[c.b];
    }
}

__global__ void __launch_bounds__(128)
ff_wmma(const float* __restrict__ W1, const float* __restrict__ b1,
        const float* __restrict__ W2, const float* __restrict__ b2,
        const float* __restrict__ w,  const float* __restrict__ meanw,
        const float* __restrict__ hs, const float* __restrict__ hss,
        const float* __restrict__ hd, const float* __restrict__ size_,
        const float* __restrict__ nf, const float* __restrict__ sA,
        const float* __restrict__ sB, const float* __restrict__ consts,
        const int* __restrict__ step, float* __restrict__ pi)
{
    // W1 as f16, column-contiguous: Wt[n][k], n<112, k<160, zero-padded.
    __shared__ __attribute__((aligned(32))) _Float16 Wt[NPAD * KPAD];
    __shared__ float b1s[NPAD], w2s[NPAD];
    __shared__ float cst[256];

    int tid = threadIdx.x;
    for (int idx = tid; idx < NPAD * KPAD; idx += 128) {
        int n = idx / KPAD, k = idx - n * KPAD;
        float v = (n < FFH && k < FFIN) ? W1[k * FFH + n] : 0.f;
        Wt[idx] = (_Float16)v;
    }
    for (int n = tid; n < NPAD; n += 128) {
        b1s[n] = (n < FFH) ? b1[n] : 0.f;
        w2s[n] = (n < FFH) ? W2[n] : 0.f;
    }
    for (int i = tid; i < 256; i += 128) cst[i] = consts[i];
    __syncthreads();

    int wave = tid >> 5;
    int lane = tid & 31;
    int half = lane >> 4;     // which half-wave (K-half / M-half per ISA layout)
    int rrow = lane & 15;
    int tile = blockIdx.x * 4 + wave;
    int m    = tile * 16 + rrow;

    FeatCtx c;
    c.w = w; c.meanw = meanw; c.hs = hs; c.hss = hss; c.hd = hd; c.size = size_;
    c.nf = nf; c.sA = sA; c.sB = sB;
    c.cP = cst; c.cQ = cst + 64; c.cQb = cst + 128; c.cC0 = cst + 192;
    c.fi = (float)step[0];
    c.b = m / U1_;
    c.u = m - c.b * U1_;
    c.node_emb = c.b * SLEN + c.u;
    c.node_inc = c.b * SLEN + (SLEN - 1);

    v8f acc[7];
    v8f zero8 = {0.f, 0.f, 0.f, 0.f, 0.f, 0.f, 0.f, 0.f};
#pragma unroll
    for (int t = 0; t < 7; ++t) acc[t] = zero8;

#pragma unroll
    for (int ks = 0; ks < 5; ++ks) {
        // A fragment (contiguous K, matching B): lanes 0-15 -> K kb..kb+15
        // (kb = ks*32); lanes 16-31 -> K offset by 16.
        int kb = ks * 32 + (half ? 16 : 0);
        v16h a;
#pragma unroll
        for (int i = 0; i < 8; ++i) a[i]     = (_Float16)feat(c, kb + i);
#pragma unroll
        for (int i = 0; i < 8; ++i) a[8 + i] = (_Float16)feat(c, kb + 8 + i);

        int kb2 = ks * 32 + half * 16;   // B: lanes 0-15 K lo-half, 16-31 K hi-half
#pragma unroll
        for (int t = 0; t < 7; ++t) {
            const v16h* bp = (const v16h*)&Wt[(t * 16 + rrow) * KPAD + kb2];
            v16h bf = *bp;               // one aligned 32B LDS load
            acc[t] = __builtin_amdgcn_wmma_f32_16x16x32_f16(
                false, a, false, bf, (short)0, acc[t], false, false);
        }
    }

    // Fused layer 2: pi[m] = b2 + sum_n relu(h[m,n] + b1[n]) * W2[n]
    // C layout: lane<16 holds (M=r, N=lane); lane>=16 holds (M=8+r, N=lane-16).
    float part[8];
#pragma unroll
    for (int r = 0; r < 8; ++r) part[r] = 0.f;
#pragma unroll
    for (int t = 0; t < 7; ++t) {
        int n = t * 16 + rrow;
        float bv = b1s[n], wv = w2s[n];
#pragma unroll
        for (int r = 0; r < 8; ++r) {
            float h = acc[t][r] + bv;
            h = h > 0.f ? h : 0.f;
            part[r] += h * wv;
        }
    }
#pragma unroll
    for (int off = 8; off >= 1; off >>= 1) {
#pragma unroll
        for (int r = 0; r < 8; ++r) part[r] += __shfl_xor(part[r], off, 16);
    }
    if (rrow == 0) {
        float bias2 = b2[0];
#pragma unroll
        for (int r = 0; r < 8; ++r)
            pi[tile * 16 + half * 8 + r] = part[r] + bias2;
    }
}

// masked log_softmax + first-occurrence argmax per row
__global__ void softmax_argmax(const float* __restrict__ pi,
                               const unsigned char* __restrict__ mask,
                               float* __restrict__ out)
{
    __shared__ float red[256];
    __shared__ int   redi[256];
    int b = blockIdx.x, tid = threadIdx.x;

    float mymax = -3.4e38f; int myarg = U1_;
    for (int u = tid; u < U1_; u += 256) {
        float v = pi[b * U1_ + u];
        if (mask[b * U1_ + u]) v = -1.0e6f;
        if (v > mymax) { mymax = v; myarg = u; }
    }
    red[tid] = mymax; redi[tid] = myarg; __syncthreads();
    for (int o = 128; o > 0; o >>= 1) {
        if (tid < o) {
            float ov = red[tid + o]; int oi = redi[tid + o];
            if (ov > red[tid] || (ov == red[tid] && oi < redi[tid])) {
                red[tid] = ov; redi[tid] = oi;
            }
        }
        __syncthreads();
    }
    float mx = red[0]; int arg = redi[0];
    __syncthreads();

    float s = 0.f;
    for (int u = tid; u < U1_; u += 256) {
        float v = pi[b * U1_ + u];
        if (mask[b * U1_ + u]) v = -1.0e6f;
        s += expf(v - mx);
    }
    red[tid] = s; __syncthreads();
    for (int o = 128; o > 0; o >>= 1) {
        if (tid < o) red[tid] += red[tid + o];
        __syncthreads();
    }
    float lse = logf(red[0]);

    for (int u = tid; u < U1_; u += 256) {
        float v = pi[b * U1_ + u];
        if (mask[b * U1_ + u]) v = -1.0e6f;
        out[b * U1_ + u] = v - mx - lse;
    }
    if (tid == 0) out[B_ * U1_ + b] = (float)arg;   // selected, as float
}

extern "C" void kernel_launch(void* const* d_in, const int* in_sizes, int n_in,
                              void* d_out, int out_size, void* d_ws, size_t ws_size,
                              hipStream_t stream) {
    (void)in_sizes; (void)n_in; (void)out_size; (void)ws_size;
    const float* nf     = (const float*)d_in[0];
    const float* wts    = (const float*)d_in[1];
    const float* w      = (const float*)d_in[2];
    const float* hs     = (const float*)d_in[3];
    const float* hss    = (const float*)d_in[4];
    const float* hd     = (const float*)d_in[5];
    const float* size_  = (const float*)d_in[6];
    const float* W_in   = (const float*)d_in[7];
    const float* b_in   = (const float*)d_in[8];
    const float* W_self = (const float*)d_in[9];
    const float* W_nbr  = (const float*)d_in[10];
    const float* b_gnn  = (const float*)d_in[11];
    const float* W1     = (const float*)d_in[12];
    const float* b1     = (const float*)d_in[13];
    const float* W2     = (const float*)d_in[14];
    const float* b2     = (const float*)d_in[15];
    const int*   esrc   = (const int*)d_in[16];
    const int*   edst   = (const int*)d_in[17];
    const unsigned char* mask = (const unsigned char*)d_in[18];
    const int*   step   = (const int*)d_in[19];

    // workspace layout (floats): consts[256] | sA[N] | sB[N] | meanw[B] | pi[B*U1]
    float* wsF    = (float*)d_ws;
    float* consts = wsF;
    float* sA     = consts + 256;
    float* sB     = sA + N_;
    float* meanw  = sB + N_;
    float* pi     = meanw + B_;

    prep_consts<<<1, 64, 0, stream>>>(W_in, b_in, W_self, W_nbr, b_gnn, consts);
    zero_ws<<<(2 * N_ + 255) / 256, 256, 0, stream>>>(sA, 2 * N_);
    row_mean<<<B_, 256, 0, stream>>>(w, meanw);
    edge_scatter<<<(E_ + 255) / 256, 256, 0, stream>>>(esrc, edst, wts, nf, sA, sB);
    ff_wmma<<<MTILES / 4, 128, 0, stream>>>(W1, b1, W2, b2, w, meanw, hs, hss, hd,
                                            size_, nf, sA, sB, consts, step, pi);
    softmax_argmax<<<B_, 256, 0, stream>>>(pi, mask, (float*)d_out);
}
